// MPGG_51754355916803
// MI455X (gfx1250) — matrix operand order repaired and checked by reading, not compiled, not run on hardware
//
#include <hip/hip_runtime.h>

typedef __attribute__((ext_vector_type(16))) _Float16 v16h;
typedef __attribute__((ext_vector_type(8)))  _Float16 v8h;
typedef __attribute__((ext_vector_type(8)))  float    v8f;
typedef _Float16 half_t;

#define NN 384
#define HH 256
#define ZZ 128
#define EDIM 128
#define HID 1024   // 4H
#define AHID 64
#define KPANEL 128
#define BROW (KPANEL + 8)   // padded LDS row (halves), 16B-aligned stride

__device__ __forceinline__ float sigm(float x) { return 1.f / (1.f + __expf(-x)); }

// ---------------- GRU scan: nodes[384][256] -------------------------------
__global__ __launch_bounds__(256) void k_gru(const float* __restrict__ z,
                                             const float* __restrict__ Wih,
                                             const float* __restrict__ Whh,
                                             const float* __restrict__ bih,
                                             const float* __restrict__ bhh,
                                             float* __restrict__ nodes) {
    __shared__ float zs[ZZ];
    __shared__ float h[HH];
    const int t = threadIdx.x;
    if (t < ZZ) zs[t] = z[t];
    h[t] = 0.f;
    __syncthreads();

    float gir = bih[t], giz = bih[HH + t], gin = bih[2 * HH + t];
    {
        const float* wr = Wih + (size_t)t * ZZ;
        const float* wz = Wih + (size_t)(HH + t) * ZZ;
        const float* wn = Wih + (size_t)(2 * HH + t) * ZZ;
        for (int k = 0; k < ZZ; ++k) {
            float zk = zs[k];
            gir += wr[k] * zk; giz += wz[k] * zk; gin += wn[k] * zk;
        }
    }
    const float* hr = Whh + (size_t)t * HH;
    const float* hz = Whh + (size_t)(HH + t) * HH;
    const float* hn = Whh + (size_t)(2 * HH + t) * HH;
    const float bhr = bhh[t], bhz = bhh[HH + t], bhn = bhh[2 * HH + t];

    for (int n = 0; n < NN; ++n) {
        float ghr = bhr, ghz = bhz, ghn = bhn;
        for (int k = 0; k < HH; ++k) {
            float hk = h[k];
            ghr += hr[k] * hk; ghz += hz[k] * hk; ghn += hn[k] * hk;
        }
        float rg = sigm(gir + ghr);
        float ug = sigm(giz + ghz);
        float ng = tanhf(gin + rg * ghn);
        float h2 = (1.f - ug) * ng + ug * h[t];
        __syncthreads();
        h[t] = h2;
        nodes[n * HH + t] = h2;
        __syncthreads();
    }
}

// ------- U = nodes@egW1[:256] + b1 ; V = nodes@egW1[256:]  (f16 out) ------
__global__ void k_uv(const float* __restrict__ nodes, const float* __restrict__ egW1,
                     const float* __restrict__ egb1, half_t* __restrict__ U,
                     half_t* __restrict__ V) {
    int idx = blockIdx.x * blockDim.x + threadIdx.x;
    if (idx >= 2 * NN * HID) return;
    int which = idx >= NN * HID;
    int rem = idx - which * NN * HID;
    int n = rem >> 10, c = rem & (HID - 1);
    const float* nd = nodes + n * HH;
    const float* w = egW1 + (which ? (size_t)HH * HID : 0) + c;
    float acc = which ? 0.f : egb1[c];
    for (int k = 0; k < HH; ++k) acc += nd[k] * w[(size_t)k * HID];
    (which ? V : U)[(size_t)n * HID + c] = (half_t)acc;
}

// ------- M0T/M1T [64][1024] f16 : M = egW2 @ attW1[:128] (transposed) -----
__global__ void k_m(const float* __restrict__ egW2, const float* __restrict__ a0W1,
                    const float* __restrict__ a1W1, half_t* __restrict__ M0T,
                    half_t* __restrict__ M1T) {
    int idx = blockIdx.x * blockDim.x + threadIdx.x;
    if (idx >= 2 * AHID * HID) return;
    int which = idx >= AHID * HID;
    int rem = idx - which * AHID * HID;
    int n = rem >> 10, k = rem & (HID - 1);
    const float* aw = which ? a1W1 : a0W1;
    const float* w2 = egW2 + (size_t)k * EDIM;
    float acc = 0.f;
    for (int j = 0; j < EDIM; ++j) acc += w2[j] * aw[j * AHID + n];
    (which ? M1T : M0T)[(size_t)n * HID + k] = (half_t)acc;
}

// ------- c0/c1 [64] = egb2 @ attW1[:128] + attb1 --------------------------
__global__ void k_c(const float* __restrict__ egb2, const float* __restrict__ a0W1,
                    const float* __restrict__ a0b1, const float* __restrict__ a1W1,
                    const float* __restrict__ a1b1, float* __restrict__ c0,
                    float* __restrict__ c1) {
    int t = threadIdx.x;
    if (t >= 2 * AHID) return;
    int which = t >> 6, n = t & (AHID - 1);
    const float* aw = which ? a1W1 : a0W1;
    float acc = (which ? a1b1 : a0b1)[n];
    for (int j = 0; j < EDIM; ++j) acc += egb2[j] * aw[j * AHID + n];
    (which ? c1 : c0)[n] = acc;
}

// ------- P[s][64] = x[s] @ attW1[128:,:] ----------------------------------
__global__ void k_p(const float* __restrict__ x, const float* __restrict__ aW1,
                    float* __restrict__ P) {
    int idx = blockIdx.x * blockDim.x + threadIdx.x;
    if (idx >= NN * AHID) return;
    int s = idx >> 6, n = idx & (AHID - 1);
    const float* xr = x + s * HH;
    float acc = 0.f;
    for (int k = 0; k < HH; ++k) acc += xr[k] * aW1[(EDIM + k) * AHID + n];
    P[idx] = acc;
}

// ------- Edge WMMA pass: att0 matrix + conv1 edge-part --------------------
// 512 threads = 16 waves; each wave = 32 edges (1 src x 32 dsts = two 16x16
// A tiles sharing every LDS-loaded B operand). B staged in 34KB LDS panels.
__global__ __launch_bounds__(512) void k_edge0(
    const half_t* __restrict__ U, const half_t* __restrict__ V,
    const half_t* __restrict__ M0T, const half_t* __restrict__ M1T,
    const float* __restrict__ c0, const float* __restrict__ c1,
    const float* __restrict__ P0, const float* __restrict__ w0,
    const float* __restrict__ ab2p, float* __restrict__ A0,
    half_t* __restrict__ ep1) {
    __shared__ half_t Bs[2 * AHID * BROW];   // [matrix][row n][K panel]

    const int tid  = threadIdx.x;
    const int job  = blockIdx.x * 16 + (tid >> 5);   // 4608 wave-jobs
    const int lane = tid & 31;
    const int s  = job / 12;
    const int d0 = (job % 12) << 5;         // 32 dsts per wave
    const int mrow = lane & 15;             // A row / B col owned by this lane
    const int hi   = lane >> 4;             // K half select
    const int kbA  = hi << 3;               // A K-base: 0 or 8
    const int kbB  = hi << 3;               // B K-base: 0 or 8
    const half_t hzero = (half_t)0.0f;

    const half_t* Urow  = U + (size_t)s * HID;
    const half_t* VrowL = V + (size_t)(d0 + mrow) * HID;
    const half_t* VrowH = V + (size_t)(d0 + 16 + mrow) * HID;

    v8f acc0[8], acc1[8];                   // [g*4+t], g=0: dsts d0.., g=1: d0+16..
    const v8f vzero = {0.f, 0.f, 0.f, 0.f, 0.f, 0.f, 0.f, 0.f};
    for (int t = 0; t < 8; ++t) { acc0[t] = vzero; acc1[t] = vzero; }

    for (int p = 0; p < HID / KPANEL; ++p) {
        __syncthreads();
        // cooperative fill of B panel: 2 matrices x 64 rows x 128 halves
        for (int i = tid; i < 2 * AHID * (KPANEL / 8); i += 512) {
            const int m = i >> 10;          // matrix select (1024 float4 each)
            const int rem = i & 1023;
            const int n = rem >> 4;         // row
            const int c = rem & 15;         // float4 within row
            const half_t* msrc = (m ? M1T : M0T) + (size_t)n * HID + p * KPANEL;
            const float4* src = (const float4*)msrc + c;
            __builtin_prefetch((const void*)((const char*)src + KPANEL * 2), 0, 1);
            *((float4*)(Bs + (m * AHID + n) * BROW) + c) = *src;
        }
        __syncthreads();

        for (int kl = 0; kl < KPANEL; kl += 32) {
            const int k = p * KPANEL + kl;
            // two A operands: hidden = relu(U[s]+V[dst]) in f16
            v8h u0  = *(const v8h*)(Urow + k + kbA);
            v8h u1  = *(const v8h*)(Urow + k + 16 + kbA);
            v8h vl0 = *(const v8h*)(VrowL + k + kbA);
            v8h vl1 = *(const v8h*)(VrowL + k + 16 + kbA);
            v8h vh0 = *(const v8h*)(VrowH + k + kbA);
            v8h vh1 = *(const v8h*)(VrowH + k + 16 + kbA);
            v8h l0 = u0 + vl0, l1 = u1 + vl1;
            v8h g0 = u0 + vh0, g1 = u1 + vh1;
#pragma unroll
            for (int i = 0; i < 8; ++i) {
                l0[i] = __builtin_fmaxf16(l0[i], hzero);
                l1[i] = __builtin_fmaxf16(l1[i], hzero);
                g0[i] = __builtin_fmaxf16(g0[i], hzero);
                g1[i] = __builtin_fmaxf16(g1[i], hzero);
            }
            v16h aLo = __builtin_shufflevector(l0, l1, 0, 1, 2, 3, 4, 5, 6, 7,
                                               8, 9, 10, 11, 12, 13, 14, 15);
            v16h aHi = __builtin_shufflevector(g0, g1, 0, 1, 2, 3, 4, 5, 6, 7,
                                               8, 9, 10, 11, 12, 13, 14, 15);
#pragma unroll
            for (int t = 0; t < 4; ++t) {
                const int n = (t << 4) + mrow;
                const half_t* b0r = Bs + n * BROW + kl + kbB;
                v8h b0l = *(const v8h*)b0r;
                v8h b0h = *(const v8h*)(b0r + 16);
                v16h b0 = __builtin_shufflevector(b0l, b0h, 0, 1, 2, 3, 4, 5, 6, 7,
                                                  8, 9, 10, 11, 12, 13, 14, 15);
                acc0[t]     = __builtin_amdgcn_wmma_f32_16x16x32_f16(
                    false, aLo, false, b0, (short)0, acc0[t], false, false);
                acc0[4 + t] = __builtin_amdgcn_wmma_f32_16x16x32_f16(
                    false, aHi, false, b0, (short)0, acc0[4 + t], false, false);
                const half_t* b1r = Bs + (AHID + n) * BROW + kl + kbB;
                v8h b1l = *(const v8h*)b1r;
                v8h b1h = *(const v8h*)(b1r + 16);
                v16h b1 = __builtin_shufflevector(b1l, b1h, 0, 1, 2, 3, 4, 5, 6, 7,
                                                  8, 9, 10, 11, 12, 13, 14, 15);
                acc1[t]     = __builtin_amdgcn_wmma_f32_16x16x32_f16(
                    false, aLo, false, b1, (short)0, acc1[t], false, false);
                acc1[4 + t] = __builtin_amdgcn_wmma_f32_16x16x32_f16(
                    false, aHi, false, b1, (short)0, acc1[4 + t], false, false);
            }
        }
    }

    // ---- epilogue: att0 = sigmoid(relu(acc0 + c0 + P0[s]) . w0 + b2) ----
    const float ab2 = ab2p[0];
    const float* P0s = P0 + s * AHID;
#pragma unroll
    for (int g = 0; g < 2; ++g) {
        const int dg = d0 + (g << 4);
        float part[8];
        for (int r = 0; r < 8; ++r) part[r] = 0.f;
        for (int t = 0; t < 4; ++t) {
            const int col = (t << 4) + mrow;
            const float cc = c0[col] + P0s[col];
            const float wv = w0[col];
            for (int r = 0; r < 8; ++r)
                part[r] += fmaxf(acc0[g * 4 + t][r] + cc, 0.f) * wv;
        }
        for (int r = 0; r < 8; ++r) {
            float sum = part[r];
            for (int off = 8; off; off >>= 1) sum += __shfl_xor(sum, off, 32);
            if (mrow == 0) {   // lanes 0 and 16 cover edges r and r+8
                const int m = r + (hi << 3);
                const int dst = dg + m;
                A0[(size_t)s * NN + dst] = (dst == s) ? 0.f : sigm(sum + ab2);
            }
        }
        // ---- store conv1 edge-part (c1 folded in) as f16 ----
        for (int t = 0; t < 4; ++t) {
            const int col = (t << 4) + mrow;
            const float cc1 = c1[col];
            for (int r = 0; r < 8; ++r) {
                const int m = r + (hi << 3);
                const int dst = dg + m;
                ep1[((size_t)(s * NN + dst)) * AHID + col] =
                    (half_t)(acc1[g * 4 + t][r] + cc1);
            }
        }
    }
}

// ------- attention pass 2 (cheap): A1 from stored edge parts --------------
__global__ void k_att1(const half_t* __restrict__ ep1, const float* __restrict__ P1,
                       const float* __restrict__ w1, const float* __restrict__ ab2p,
                       float* __restrict__ A1) {
    int e = blockIdx.x * blockDim.x + threadIdx.x;
    if (e >= NN * NN) return;
    int s = e / NN, d = e - s * NN;
    const half_t* ep = ep1 + (size_t)e * AHID;
    const float* P1s = P1 + s * AHID;
    float acc = 0.f;
    for (int j = 0; j < AHID; ++j)
        acc += fmaxf((float)ep[j] + P1s[j], 0.f) * w1[j];
    A1[e] = (s == d) ? 0.f : sigm(acc + ab2p[0]);
}

// ------- agg[d][j] = sum_s A[s][d] * X[s][j] ------------------------------
__global__ void k_agg(const float* __restrict__ A, const float* __restrict__ X,
                      float* __restrict__ out) {
    int idx = blockIdx.x * blockDim.x + threadIdx.x;
    if (idx >= NN * HH) return;
    int d = idx >> 8, j = idx & (HH - 1);
    float acc = 0.f;
    for (int s = 0; s < NN; ++s) acc += A[(size_t)s * NN + d] * X[(size_t)s * HH + j];
    out[idx] = acc;
}

// ------- y = act(in @ W + b), W [256][256] --------------------------------
__global__ void k_lin(const float* __restrict__ in, const float* __restrict__ W,
                      const float* __restrict__ b, float* __restrict__ out, int do_relu) {
    int idx = blockIdx.x * blockDim.x + threadIdx.x;
    if (idx >= NN * HH) return;
    int n = idx >> 8, j = idx & (HH - 1);
    const float* ir = in + (size_t)n * HH;
    float acc = b[j];
    for (int k = 0; k < HH; ++k) acc += ir[k] * W[k * HH + j];
    out[idx] = do_relu ? fmaxf(acc, 0.f) : acc;
}

// ------- final: out = ((rowsum A1) @ x) @ W + N*b -------------------------
__global__ __launch_bounds__(384) void k_final(const float* __restrict__ A1,
                                               const float* __restrict__ x,
                                               const float* __restrict__ W,
                                               const float* __restrict__ b,
                                               float* __restrict__ out) {
    __shared__ float r[NN];
    __shared__ float tk[HH];
    const int t = threadIdx.x;
    {
        float acc = 0.f;
        const float* row = A1 + (size_t)t * NN;
        for (int d = 0; d < NN; ++d) acc += row[d];
        r[t] = acc;
    }
    __syncthreads();
    if (t < HH) {
        float acc = 0.f;
        for (int s = 0; s < NN; ++s) acc += r[s] * x[(size_t)s * HH + t];
        tk[t] = acc;
    }
    __syncthreads();
    if (t < HH) {
        float acc = 0.f;
        for (int k = 0; k < HH; ++k) acc += tk[k] * W[k * HH + t];
        out[t] = acc + (float)NN * b[t];
    }
}

extern "C" void kernel_launch(void* const* d_in, const int* in_sizes, int n_in,
                              void* d_out, int out_size, void* d_ws, size_t ws_size,
                              hipStream_t stream) {
    (void)in_sizes; (void)n_in; (void)out_size; (void)ws_size;
    const float* z       = (const float*)d_in[0];
    const float* gru_Wih = (const float*)d_in[1];
    const float* gru_Whh = (const float*)d_in[2];
    const float* gru_bih = (const float*)d_in[3];
    const float* gru_bhh = (const float*)d_in[4];
    const float* eg_W1   = (const float*)d_in[5];
    const float* eg_b1   = (const float*)d_in[6];
    const float* eg_W2   = (const float*)d_in[7];
    const float* eg_b2   = (const float*)d_in[8];
    const float* a0_W1   = (const float*)d_in[9];
    const float* a0_b1   = (const float*)d_in[10];
    const float* a0_W2   = (const float*)d_in[11];
    const float* a0_b2   = (const float*)d_in[12];
    const float* a1_W1   = (const float*)d_in[13];
    const float* a1_b1   = (const float*)d_in[14];
    const float* a1_W2   = (const float*)d_in[15];
    const float* a1_b2   = (const float*)d_in[16];
    const float* c0_W    = (const float*)d_in[17];
    const float* c0_b    = (const float*)d_in[18];
    const float* c1_W    = (const float*)d_in[19];
    const float* c1_b    = (const float*)d_in[20];
    float* out = (float*)d_out;

    char* p = (char*)d_ws;
    auto alloc = [&p](size_t bytes) -> void* {
        void* r = (void*)p;
        p += (bytes + 255) & ~(size_t)255;
        return r;
    };
    float*  nodes = (float*)alloc((size_t)NN * HH * 4);
    half_t* U     = (half_t*)alloc((size_t)NN * HID * 2);
    half_t* V     = (half_t*)alloc((size_t)NN * HID * 2);
    half_t* M0T   = (half_t*)alloc((size_t)AHID * HID * 2);
    half_t* M1T   = (half_t*)alloc((size_t)AHID * HID * 2);
    float*  c0    = (float*)alloc(AHID * 4);
    float*  c1    = (float*)alloc(AHID * 4);
    float*  P0    = (float*)alloc((size_t)NN * AHID * 4);
    float*  P1    = (float*)alloc((size_t)NN * AHID * 4);
    float*  A0    = (float*)alloc((size_t)NN * NN * 4);
    float*  A1    = (float*)alloc((size_t)NN * NN * 4);
    float*  agg0  = (float*)alloc((size_t)NN * HH * 4);
    float*  xbuf  = (float*)alloc((size_t)NN * HH * 4);
    half_t* ep1   = (half_t*)alloc((size_t)NN * NN * AHID * 2);

    k_gru<<<1, 256, 0, stream>>>(z, gru_Wih, gru_Whh, gru_bih, gru_bhh, nodes);
    k_uv<<<(2 * NN * HID) / 256, 256, 0, stream>>>(nodes, eg_W1, eg_b1, U, V);
    k_m<<<(2 * AHID * HID) / 256, 256, 0, stream>>>(eg_W2, a0_W1, a1_W1, M0T, M1T);
    k_c<<<1, 128, 0, stream>>>(eg_b2, a0_W1, a0_b1, a1_W1, a1_b1, c0, c1);
    k_p<<<(NN * AHID) / 256, 256, 0, stream>>>(nodes, a0_W1, P0);
    // 4608 dual-tile wave-jobs, 16 waves per block -> 288 blocks
    k_edge0<<<288, 512, 0, stream>>>(U, V, M0T, M1T, c0, c1, P0, a0_W2, a0_b2, A0, ep1);
    k_agg<<<(NN * HH) / 256, 256, 0, stream>>>(A0, nodes, agg0);
    k_lin<<<(NN * HH) / 256, 256, 0, stream>>>(agg0, c0_W, c0_b, xbuf, 1);
    k_p<<<(NN * AHID) / 256, 256, 0, stream>>>(xbuf, a1_W1, P1);
    k_att1<<<(NN * NN) / 256, 256, 0, stream>>>(ep1, P1, a1_W2, a1_b2, A1);
    k_final<<<1, 384, 0, stream>>>(A1, xbuf, c1_W, c1_b, out);
}
